// DotProductAttention_34067680592084
// MI455X (gfx1250) — matrix operand order repaired and checked by reading, not compiled, not run on hardware
//
#include <hip/hip_runtime.h>

// ---------------------------------------------------------------------------
// Flash attention with tanh logit clipping for MI455X (gfx1250, wave32).
// logits = 10*tanh((Q K^T)/sqrt(128)), masked keys -> -inf, softmax, P V.
// bf16 WMMA (16x16x32) with fp32 accumulation, online softmax.
// Each wave owns TWO 16-row query tiles -> 64 WMMA per key block per wave
// against one set of K/V B-fragments (2x arithmetic intensity on LDS).
// ---------------------------------------------------------------------------

typedef __attribute__((ext_vector_type(16))) __bf16 v16bf;
typedef __attribute__((ext_vector_type(8)))  __bf16 v8bf;
typedef __attribute__((ext_vector_type(4)))  __bf16 v4bf;
typedef __attribute__((ext_vector_type(2)))  __bf16 v2bf;
typedef __attribute__((ext_vector_type(8)))  float  v8f;

#define B_dim   16
#define M_dim   2048
#define N_dim   2048
#define D_dim   128
#define BM      128         // queries per workgroup (4 waves x 2 tiles x 16 rows)
#define BN      64          // keys per block iteration
#define NTHREADS 128
#define CLIPC   10.0f
#define SCALE   0.08838834764831845f   // 1/sqrt(128)
#define LOG2E   1.4426950408889634f

__device__ __forceinline__ float fast_tanh(float x) {
#if __has_builtin(__builtin_amdgcn_tanhf)
    return __builtin_amdgcn_tanhf(x);
#else
    // tanh(x) = 1 - 2/(exp(2x)+1), exact limits at +-inf, via v_exp/v_rcp
    float e = __builtin_amdgcn_exp2f(x * (2.0f * LOG2E));
    return 1.0f - 2.0f * __builtin_amdgcn_rcpf(e + 1.0f);
#endif
}

// 16 contiguous bf16 from LDS (16B-aligned) -> v16bf
__device__ __forceinline__ v16bf ld16(const __bf16* p) {
    v8bf lo = *(const v8bf*)p;
    v8bf hi = *(const v8bf*)(p + 8);
    return __builtin_shufflevector(lo, hi, 0,1,2,3,4,5,6,7,8,9,10,11,12,13,14,15);
}
// two separate 8-element bf16 chunks -> v16bf
__device__ __forceinline__ v16bf ld8x2(const __bf16* p0, const __bf16* p1) {
    v8bf lo = *(const v8bf*)p0;
    v8bf hi = *(const v8bf*)p1;
    return __builtin_shufflevector(lo, hi, 0,1,2,3,4,5,6,7,8,9,10,11,12,13,14,15);
}

__global__ __launch_bounds__(NTHREADS)
void DotProductAttention_tanh_flash_kernel(const float* __restrict__ Qp,
                                           const float* __restrict__ Kp,
                                           const float* __restrict__ Vp,
                                           const unsigned char* __restrict__ maskp,
                                           float* __restrict__ Op)
{
    // LDS: K natural [key][d] bf16, V transposed [d][key] bf16 (contraction dim
    // must be lane-contiguous for WMMA fragments), mask bias, per-wave P tiles.
    __shared__ __align__(16) __bf16 sK [BN][D_dim];     // 16 KB
    __shared__ __align__(16) __bf16 sVt[D_dim][BN];     // 16 KB
    __shared__ __align__(16) __bf16 sP [4][32][BN];     // 16 KB (per-wave scratch)
    __shared__ float sBias[BN];

    const int tid   = threadIdx.x;
    const int wave  = tid >> 5;
    const int lane  = tid & 31;
    const int ln    = lane & 15;     // column / row-within-16 role
    const int kh    = lane >> 4;     // K-half selector in fragment layouts
    const int batch = blockIdx.y;
    const int qbase = blockIdx.x * BM;
    const int qrow0 = qbase + wave * 32;   // this wave's first query row (2 tiles)

    // uniform per-batch scalar bases; all varying offsets are 32-bit ints so
    // global accesses lower to GVS mode (saddr + 32-bit voffset, no v_add_u64)
    const float*         kBase = Kp + (size_t)batch * (N_dim * D_dim);
    const float*         vBase = Vp + (size_t)batch * (N_dim * D_dim);
    const float*         qBase = Qp + (size_t)batch * (M_dim * D_dim);
    const unsigned char* mBase = maskp + (size_t)batch * N_dim;
    float*               oBase = Op + (size_t)batch * (M_dim * D_dim);

    // ---- Load two 16x128 Q tiles as persistent bf16 A-fragments ----
    // A-matrix 16x32 bf16 layout (ISA 7.12.2): lane m=ln (both halves are rows
    // 0..15). elems 0..7 -> K = 32*kk + 8*kh + e ; elems 8..15 -> +16.
    v16bf qa[2][4];
#pragma unroll
    for (int t = 0; t < 2; ++t) {
        const int qoff = (qrow0 + t * 16 + ln) * D_dim;
#pragma unroll
        for (int kk = 0; kk < 4; ++kk) {
            const int c0 = qoff + 32 * kk + 8 * kh;
            float4 f0 = *(const float4*)(qBase + c0);
            float4 f1 = *(const float4*)(qBase + c0 + 4);
            float4 f2 = *(const float4*)(qBase + c0 + 16);
            float4 f3 = *(const float4*)(qBase + c0 + 20);
            v16bf q;
            q[0]=(__bf16)f0.x;  q[1]=(__bf16)f0.y;  q[2]=(__bf16)f0.z;  q[3]=(__bf16)f0.w;
            q[4]=(__bf16)f1.x;  q[5]=(__bf16)f1.y;  q[6]=(__bf16)f1.z;  q[7]=(__bf16)f1.w;
            q[8]=(__bf16)f2.x;  q[9]=(__bf16)f2.y;  q[10]=(__bf16)f2.z; q[11]=(__bf16)f2.w;
            q[12]=(__bf16)f3.x; q[13]=(__bf16)f3.y; q[14]=(__bf16)f3.z; q[15]=(__bf16)f3.w;
            qa[t][kk] = q;
        }
    }

    // ---- Online-softmax running state (rows m = r + 8*kh per tile) ----
    float mrun[2][8], lrun[2][8];
    v8f acc[2][8];                   // output tiles: 2 x (16 rows x 128 dims)
#pragma unroll
    for (int t = 0; t < 2; ++t)
#pragma unroll
        for (int r = 0; r < 8; ++r) { mrun[t][r] = -__builtin_inff(); lrun[t][r] = 0.0f; }
#pragma unroll
    for (int t = 0; t < 2; ++t)
#pragma unroll
        for (int dt = 0; dt < 8; ++dt)
#pragma unroll
            for (int r = 0; r < 8; ++r) acc[t][dt][r] = 0.0f;

    // ======================= key-block loop =======================
#pragma unroll 1
    for (int kb = 0; kb < N_dim; kb += BN) {

        // ---- cooperative staging: global fp32 -> LDS bf16 ----
        // K tile, natural layout; packed 4xbf16 (b64) LDS stores
#pragma unroll
        for (int i = 0; i < (BN * D_dim / 4) / NTHREADS; ++i) {
            int idx = (tid + i * NTHREADS) * 4;
            int key = idx >> 7;            // /128
            int d   = idx & (D_dim - 1);
            float4 f = *(const float4*)(kBase + kb * D_dim + idx);
            v4bf kv;
            kv[0]=(__bf16)f.x; kv[1]=(__bf16)f.y; kv[2]=(__bf16)f.z; kv[3]=(__bf16)f.w;
            *(v4bf*)&sK[key][d] = kv;
        }
        // V tile, transposed; key-pairs so stores are packed 2xbf16 (b32)
#pragma unroll
        for (int i = 0; i < (BN / 2) * (D_dim / 4) / NTHREADS; ++i) {
            int w   = tid + i * NTHREADS;        // 0..1023
            int kp2 = (w >> 5) << 1;             // even key
            int d   = (w & 31) * 4;
            int off = kb * D_dim + kp2 * D_dim + d;
            float4 f0 = *(const float4*)(vBase + off);
            float4 f1 = *(const float4*)(vBase + off + D_dim);
#pragma unroll
            for (int j = 0; j < 4; ++j) {
                float a = (j==0)?f0.x:(j==1)?f0.y:(j==2)?f0.z:f0.w;
                float b = (j==0)?f1.x:(j==1)?f1.y:(j==2)?f1.z:f1.w;
                v2bf pv; pv[0]=(__bf16)a; pv[1]=(__bf16)b;
                *(v2bf*)&sVt[d + j][kp2] = pv;
            }
        }
        if (tid < BN)
            sBias[tid] = mBase[kb + tid] ? -__builtin_inff() : 0.0f;
        __syncthreads();

        // ---- S = Q K^T : 4 key sub-tiles x 4 K-steps, 2 query tiles ----
        v8f s[2][4];
#pragma unroll
        for (int j = 0; j < 4; ++j) {
            v8f s0, s1;
#pragma unroll
            for (int r = 0; r < 8; ++r) { s0[r] = 0.0f; s1[r] = 0.0f; }
#pragma unroll
            for (int kk = 0; kk < 4; ++kk) {
                // B-matrix 32x16 bf16: lane holds column n=ln (key 16*j+ln),
                // dims 32*kk + 8*kh + e / +16 (matches A layout)
                v16bf bfrag = ld8x2(&sK[16 * j + ln][32 * kk + 8 * kh],
                                    &sK[16 * j + ln][32 * kk + 16 + 8 * kh]);
                s0 = __builtin_amdgcn_wmma_f32_16x16x32_bf16(
                        false, qa[0][kk], false, bfrag, (short)0, s0, false, false);
                s1 = __builtin_amdgcn_wmma_f32_16x16x32_bf16(
                        false, qa[1][kk], false, bfrag, (short)0, s1, false, false);
            }
            s[0][j] = s0; s[1][j] = s1;
        }

        // ---- per-tile: tanh-clip + mask bias, online softmax, P -> LDS ----
#pragma unroll
        for (int t = 0; t < 2; ++t) {
            float pcl[4][8];
            float rmax[8];
#pragma unroll
            for (int r = 0; r < 8; ++r) rmax[r] = -__builtin_inff();
#pragma unroll
            for (int j = 0; j < 4; ++j) {
                float bias = sBias[16 * j + ln];
#pragma unroll
                for (int r = 0; r < 8; ++r) {
                    float x = CLIPC * fast_tanh(s[t][j][r] * SCALE) + bias;
                    pcl[j][r] = x;
                    rmax[r] = fmaxf(rmax[r], x);
                }
            }
            // row reduction across 16 lanes (halves own disjoint row sets)
#pragma unroll
            for (int msk = 1; msk < 16; msk <<= 1)
#pragma unroll
                for (int r = 0; r < 8; ++r)
                    rmax[r] = fmaxf(rmax[r], __shfl_xor(rmax[r], msk, 32));

            float mnew[8], scl[8], rsum[8];
#pragma unroll
            for (int r = 0; r < 8; ++r) {
                mnew[r] = fmaxf(fmaxf(mrun[t][r], rmax[r]), -20.0f); // no all-masked NaN
                scl[r]  = __builtin_amdgcn_exp2f((mrun[t][r] - mnew[r]) * LOG2E);
                rsum[r] = 0.0f;
            }
#pragma unroll
            for (int j = 0; j < 4; ++j)
#pragma unroll
                for (int r = 0; r < 8; ++r) {
                    float p = __builtin_amdgcn_exp2f((pcl[j][r] - mnew[r]) * LOG2E);
                    pcl[j][r] = p;
                    rsum[r] += p;
                }
#pragma unroll
            for (int msk = 1; msk < 16; msk <<= 1)
#pragma unroll
                for (int r = 0; r < 8; ++r)
                    rsum[r] += __shfl_xor(rsum[r], msk, 32);
#pragma unroll
            for (int r = 0; r < 8; ++r) {
                lrun[t][r] = lrun[t][r] * scl[r] + rsum[r];
                mrun[t][r] = mnew[r];
            }
            // rescale accumulator
#pragma unroll
            for (int dt = 0; dt < 8; ++dt)
#pragma unroll
                for (int r = 0; r < 8; ++r) acc[t][dt][r] *= scl[r];

            // P (C-layout) -> per-wave LDS [row][key] for A-fragment reads
#pragma unroll
            for (int j = 0; j < 4; ++j) {
                int n = 16 * j + ln;
#pragma unroll
                for (int r = 0; r < 8; ++r)
                    sP[wave][t * 16 + r + 8 * kh][n] = (__bf16)pcl[j][r];
            }
        }
        // same-wave LDS ordering: DS ops in-order, compiler inserts s_wait_dscnt

        // ---- O += P V : 2 K-steps x 8 dim tiles x 2 query tiles ----
#pragma unroll
        for (int kk = 0; kk < 2; ++kk) {
            // A-matrix 16x32: lane row m=ln; keys 32*kk+8*kh+e / +16
            v16bf pa0 = ld8x2(&sP[wave][ln]     [32 * kk + 8 * kh],
                              &sP[wave][ln]     [32 * kk + 16 + 8 * kh]);
            v16bf pa1 = ld8x2(&sP[wave][16 + ln][32 * kk + 8 * kh],
                              &sP[wave][16 + ln][32 * kk + 16 + 8 * kh]);
#pragma unroll
            for (int dt = 0; dt < 8; ++dt) {
                // B-matrix 32x16: lane column n=ln (dim 16*dt+ln), keys
                // 32*kk + 8*kh + e / +16 in transposed V
                v16bf vb = ld8x2(&sVt[16 * dt + ln][32 * kk + 8 * kh],
                                 &sVt[16 * dt + ln][32 * kk + 16 + 8 * kh]);
                acc[0][dt] = __builtin_amdgcn_wmma_f32_16x16x32_bf16(
                                false, pa0, false, vb, (short)0, acc[0][dt], false, false);
                acc[1][dt] = __builtin_amdgcn_wmma_f32_16x16x32_bf16(
                                false, pa1, false, vb, (short)0, acc[1][dt], false, false);
            }
        }
        __syncthreads();   // everyone done with sK/sVt before restaging
    }

    // ---- normalize and store ----
#pragma unroll
    for (int t = 0; t < 2; ++t) {
        float rcp[8];
#pragma unroll
        for (int r = 0; r < 8; ++r) rcp[r] = __builtin_amdgcn_rcpf(lrun[t][r]);
#pragma unroll
        for (int dt = 0; dt < 8; ++dt) {
#pragma unroll
            for (int r = 0; r < 8; ++r) {
                int row = qrow0 + t * 16 + r + 8 * kh;
                int col = 16 * dt + ln;
                oBase[row * D_dim + col] = acc[t][dt][r] * rcp[r];
            }
        }
    }
}

extern "C" void kernel_launch(void* const* d_in, const int* in_sizes, int n_in,
                              void* d_out, int out_size, void* d_ws, size_t ws_size,
                              hipStream_t stream) {
    const float* Q = (const float*)d_in[0];
    const float* K = (const float*)d_in[1];
    const float* V = (const float*)d_in[2];
    const unsigned char* mask = (const unsigned char*)d_in[3];
    float* out = (float*)d_out;
    (void)in_sizes; (void)n_in; (void)ws_size; (void)d_ws; (void)out_size;

    dim3 grid(M_dim / BM, B_dim);
    dim3 block(NTHREADS);
    DotProductAttention_tanh_flash_kernel<<<grid, block, 0, stream>>>(Q, K, V, mask, out);
}
